// CenterLoss_47313359733236
// MI455X (gfx1250) — matrix operand order repaired and checked by reading, not compiled, not run on hardware
//
#include <hip/hip_runtime.h>

typedef __attribute__((ext_vector_type(16))) _Float16 v16h;
typedef __attribute__((ext_vector_type(8)))  float    v8f;

#define N_ROWS    65536
#define N_CLS     1000
#define FEAT      256
#define C_TILES   63      // ceil(1000/16)
#define BLOCK_I   64      // rows per block (4 waves * 16 rows)
#define THREADS   128
#define LDS_PITCH (FEAT + 8)   // halves; 528B row stride => conflict-free b128

union F4x2 { float4 v[2]; float f[8]; };

// Pre-pass: c_sq[c] = ||centers[c]||^2 into workspace; also zero the output scalar.
__global__ void CenterLoss_csq_kernel(const float* __restrict__ centers,
                                      float* __restrict__ csq,
                                      float* __restrict__ out) {
    int t = blockIdx.x * blockDim.x + threadIdx.x;
    if (t == 0) out[0] = 0.0f;
    if (t < N_CLS) {
        const float* r = centers + (size_t)t * FEAT;
        float s = 0.0f;
#pragma unroll 8
        for (int d = 0; d < FEAT; ++d) { float v = r[d]; s += v * v; }
        csq[t] = s;
    }
}

// Prefetch one 16x256 centers tile (f32) into registers; rows clamped in-bounds.
__device__ __forceinline__ void stage_load(const float* __restrict__ centers,
                                           int ct, int tid,
                                           float4 regs[8], int valid[8]) {
    const int c0 = ct * 16;
#pragma unroll
    for (int j = 0; j < 8; ++j) {
        int lin  = tid + THREADS * j;      // 0..1023 float4 slots
        int cr   = lin >> 6;               // tile row 0..15
        int c4   = lin & 63;               // float4 column
        int cls  = c0 + cr;
        int clsc = (cls < N_CLS) ? cls : (N_CLS - 1);
        valid[j] = (cls < N_CLS);
        regs[j]  = ((const float4*)(centers + (size_t)clsc * FEAT))[c4];
    }
}

// Convert staged tile to f16 (zeroing padded classes) and store to an LDS buffer.
__device__ __forceinline__ void stage_store(_Float16 (*buf)[LDS_PITCH],
                                            int tid, const float4 regs[8],
                                            const int valid[8]) {
#pragma unroll
    for (int j = 0; j < 8; ++j) {
        int lin = tid + THREADS * j;
        int cr  = lin >> 6;
        int c4  = lin & 63;
        float s = valid[j] ? 1.0f : 0.0f;
        float4 v = regs[j];
        union { _Float16 h[4]; unsigned long long u; } p;
        p.h[0] = (_Float16)(v.x * s);
        p.h[1] = (_Float16)(v.y * s);
        p.h[2] = (_Float16)(v.z * s);
        p.h[3] = (_Float16)(v.w * s);
        *(unsigned long long*)&buf[cr][c4 * 4] = p.u;   // 8B aligned (528%8==0)
    }
}

__global__ __launch_bounds__(THREADS)
void CenterLoss_main_kernel(const int*   __restrict__ gt,
                            const float* __restrict__ feat,
                            const float* __restrict__ centers,
                            const float* __restrict__ csq,
                            float*       __restrict__ out) {
    __shared__ _Float16 cen[2][16][LDS_PITCH];   // double-buffered centers tile
    __shared__ float    blk_sum;

    const int tid  = threadIdx.x;
    const int wave = tid >> 5;
    const int lane = tid & 31;
    const int n    = lane & 15;   // N index (class within tile) / A row M
    const int g    = lane >> 4;   // lane group (K-half selector)

    const int row0  = blockIdx.x * BLOCK_I + wave * 16; // wave's 16-row M tile
    const int myrow = row0 + n;                         // A-matrix row for this lane

    // ---- Load A fragments once (features row -> f16, ISA 16-bit A layout),
    //      accumulating this lane's partial sum-of-squares in f32. ----
    v16h  A[8];
    float fsq_part = 0.0f;
    const float* frow = feat + (size_t)myrow * FEAT;
#pragma unroll
    for (int kk = 0; kk < 8; ++kk) {
        // lane group g holds K = 32kk + [8g, 8g+8) and 32kk + 16 + [8g, 8g+8)
        const float4* p0 = (const float4*)(frow + 32 * kk + 8 * g);
        const float4* p1 = (const float4*)(frow + 32 * kk + 16 + 8 * g);
        F4x2 lo, hi;
        lo.v[0] = p0[0]; lo.v[1] = p0[1];
        hi.v[0] = p1[0]; hi.v[1] = p1[1];
        v16h a;
#pragma unroll
        for (int e = 0; e < 8; ++e) {
            float x = lo.f[e], y = hi.f[e];
            fsq_part += x * x + y * y;
            a[e]     = (_Float16)x;
            a[8 + e] = (_Float16)y;
        }
        A[kk] = a;
    }
    // Row m's squares are split across lanes m and m+16: combine.
    float fsq_full = fsq_part + __shfl_xor(fsq_part, 16, 32);

    float cross_acc = 0.0f;   // sum of mask * (f.c)
    float csq_acc   = 0.0f;   // sum of mask * c_sq
    int   cnt[8];             // per (lane, r): # masked entries in row M(r)
#pragma unroll
    for (int r = 0; r < 8; ++r) cnt[r] = 0;

    // ---- Prologue: stage tile 0 into LDS buffer 0 ----
    float4 sregs[8]; int svalid[8];
    stage_load(centers, 0, tid, sregs, svalid);
    stage_store(cen[0], tid, sregs, svalid);
    __syncthreads();

    for (int ct = 0; ct < C_TILES; ++ct) {
        const int cur    = ct & 1;
        const int c0     = ct * 16;
        const int col    = c0 + n;
        const int cvalid = (col < N_CLS) ? 1 : 0;
        const int colc   = cvalid ? col : (N_CLS - 1);

        // Issue this tile's gt mask loads unconditionally (clamped column),
        // batched so they overlap the WMMAs below. C/D layout: M = r + 8g.
        int gvs[8];
        const size_t gbase = (size_t)(row0 + 8 * g) * N_CLS + (size_t)colc;
#pragma unroll
        for (int r = 0; r < 8; ++r) gvs[r] = gt[gbase + (size_t)r * N_CLS];
        const float cs = csq[colc];

        // Prefetch next centers tile from global into registers.
        if (ct + 1 < C_TILES) stage_load(centers, ct + 1, tid, sregs, svalid);

        // ---- 16x16 cross tile via 8 chained WMMAs over K=256 ----
        v8f acc = {0.f, 0.f, 0.f, 0.f, 0.f, 0.f, 0.f, 0.f};
#pragma unroll
        for (int kk = 0; kk < 8; ++kk) {
            // B layout: lane = n + 16g holds column n, K = 32kk + [8g, 8g+8) and 32kk + 16 + [8g, 8g+8)
            const _Float16* bp = &cen[cur][n][32 * kk + 8 * g];
            v16h b;
#pragma unroll
            for (int e = 0; e < 16; ++e) b[e] = bp[e < 8 ? e : e + 8];
            acc = __builtin_amdgcn_wmma_f32_16x16x32_f16(
                      false, A[kk], false, b, (short)0, acc, false, false);
        }

        // ---- Mask + accumulate ----
        int tcnt = 0;
#pragma unroll
        for (int r = 0; r < 8; ++r) {
            int mk = cvalid & ((gvs[r] > 0) ? 1 : 0);
            cnt[r]    += mk;
            tcnt      += mk;
            cross_acc += mk ? acc[r] : 0.0f;
        }
        csq_acc += (cvalid ? cs : 0.0f) * (float)tcnt;

        // Write next tile into the other LDS buffer, then one barrier.
        if (ct + 1 < C_TILES) stage_store(cen[cur ^ 1], tid, sregs, svalid);
        __syncthreads();
    }

    // ---- f_sq term: fetch f_sq for rows 8g..8g+7 (held in lanes 8g+r). ----
    float fsq_term = 0.0f;
#pragma unroll
    for (int r = 0; r < 8; ++r) {
        float fr = __shfl(fsq_full, 8 * g + r, 32);
        fsq_term += fr * (float)cnt[r];
    }

    float total = fsq_term + csq_acc - 2.0f * cross_acc;

    // ---- Wave reduce, then block reduce, then one global atomic. ----
#pragma unroll
    for (int off = 16; off > 0; off >>= 1)
        total += __shfl_xor(total, off, 32);

    if (tid == 0) blk_sum = 0.0f;
    __syncthreads();
    if (lane == 0) atomicAdd(&blk_sum, total);
    __syncthreads();
    if (tid == 0) atomicAdd(out, blk_sum * (1.0f / (float)N_ROWS));
}

extern "C" void kernel_launch(void* const* d_in, const int* in_sizes, int n_in,
                              void* d_out, int out_size, void* d_ws, size_t ws_size,
                              hipStream_t stream) {
    const int*   gt       = (const int*)  d_in[0];   // [N, 1000] int32
    const float* features = (const float*)d_in[1];   // [N, 256]  f32
    const float* centers  = (const float*)d_in[2];   // [1000, 256] f32
    float*       out      = (float*)d_out;           // scalar f32
    float*       csq      = (float*)d_ws;            // 1000 floats of scratch

    CenterLoss_csq_kernel<<<4, 256, 0, stream>>>(centers, csq, out);
    CenterLoss_main_kernel<<<N_ROWS / BLOCK_I, THREADS, 0, stream>>>(
        gt, features, centers, csq, out);
}
